// Attention_74139725463561
// MI455X (gfx1250) — matrix operand order repaired and checked by reading, not compiled, not run on hardware
//
#include <hip/hip_runtime.h>

typedef __attribute__((ext_vector_type(16))) __bf16 v16bf;
typedef __attribute__((ext_vector_type(8)))  float  v8f;

static constexpr int CB = 4, CS = 2048, CD = 512, CH = 8, CDH = 64, CINNER = 512;

union Frag { v16bf v; uint4 u[2]; };

static __device__ __forceinline__ unsigned pack2(float a, float b) {
  unsigned short x = __builtin_bit_cast(unsigned short, (__bf16)a);
  unsigned short y = __builtin_bit_cast(unsigned short, (__bf16)b);
  return (unsigned)x | ((unsigned)y << 16);
}

// Async global -> LDS copy of 16 bytes (GV addressing), tracked by ASYNCcnt.
static __device__ __forceinline__ void async_copy_b128(const void* gptr, void* lptr) {
  unsigned lds = (unsigned)(unsigned long long)lptr;  // low 32 bits = LDS offset
  asm volatile("global_load_async_to_lds_b128 %0, %1, off"
               :: "v"(lds), "v"(gptr)
               : "memory");
}
static __device__ __forceinline__ void async_wait0() {
  asm volatile("s_wait_asynccnt 0x0" ::: "memory");
}
// Wait until at most the 8 most recently issued async copies remain in flight
// (async loads complete in order, so everything older is then done).
static __device__ __forceinline__ void async_wait8() {
  asm volatile("s_wait_asynccnt 0x8" ::: "memory");
}

// A-matrix fragment: LDS row-major [M][K], 16x32 bf16 tile.
static __device__ __forceinline__ v16bf lds_load_a(const __bf16* base, int ld, int k0, int lane) {
  int m = lane & 15, h = lane >> 4;
  Frag f;
  f.u[0] = *(const uint4*)(base + m * ld + k0 + 8 * h);
  f.u[1] = *(const uint4*)(base + m * ld + k0 + 16 + 8 * h);
  return f.v;
}
// B-matrix fragment from transposed (N-major) LDS [N][K], 32x16 bf16 tile.
static __device__ __forceinline__ v16bf lds_load_bt(const __bf16* base, int ld, int k0, int lane) {
  int n = lane & 15, h = lane >> 4;
  Frag f;
  f.u[0] = *(const uint4*)(base + n * ld + k0 + 8 * h);
  f.u[1] = *(const uint4*)(base + n * ld + k0 + 16 + 8 * h);
  return f.v;
}

#define WMMA_BF16(a, b, c) \
  __builtin_amdgcn_wmma_f32_16x16x32_bf16(false, (a), false, (b), (short)0, (c), false, false)

// ---------------------------------------------------------------------------
// Kernel 1: qkv = x @ W_qkv ; split into Q (scaled by 1/8), K, V(transposed)
// ---------------------------------------------------------------------------
__global__ __launch_bounds__(128) void qkv_proj_kernel(
    const float* __restrict__ x, const float* __restrict__ Wqkv,
    __bf16* __restrict__ qbuf, __bf16* __restrict__ kbuf, __bf16* __restrict__ vbuf) {
  __shared__ __align__(16) __bf16 As[64 * 40];
  __shared__ __align__(16) __bf16 Bs[64 * 40];
  const int tid = threadIdx.x, lane = tid & 31, wave = tid >> 5;
  const int nbase = blockIdx.x * 64, mbase = blockIdx.y * 64;

  const v8f vzero = {0, 0, 0, 0, 0, 0, 0, 0};
  v8f acc[4];
  #pragma unroll
  for (int c = 0; c < 4; ++c) acc[c] = vzero;

  for (int kc = 0; kc < CD / 32; ++kc) {
    const int k0 = kc * 32;
    __syncthreads();
    // A tile: 64 rows x 32 K, fp32 -> bf16
    #pragma unroll
    for (int i = 0; i < 4; ++i) {
      int f4 = tid + i * 128;
      int row = f4 >> 3, c4 = (f4 & 7) * 4;
      float4 v = *(const float4*)(x + (size_t)(mbase + row) * CD + k0 + c4);
      uint2 p; p.x = pack2(v.x, v.y); p.y = pack2(v.z, v.w);
      *(uint2*)&As[row * 40 + c4] = p;
    }
    // B tile: 32 K-rows x 64 N-cols of W_qkv, stored transposed [n][k]
    #pragma unroll
    for (int i = 0; i < 4; ++i) {
      int f4 = tid + i * 128;
      int kr = f4 >> 4, n4 = (f4 & 15) * 4;
      float4 v = *(const float4*)(Wqkv + (size_t)(k0 + kr) * (3 * CINNER) + nbase + n4);
      Bs[(n4 + 0) * 40 + kr] = (__bf16)v.x;
      Bs[(n4 + 1) * 40 + kr] = (__bf16)v.y;
      Bs[(n4 + 2) * 40 + kr] = (__bf16)v.z;
      Bs[(n4 + 3) * 40 + kr] = (__bf16)v.w;
    }
    __syncthreads();
    v16bf a = lds_load_a(As + wave * 16 * 40, 40, 0, lane);
    #pragma unroll
    for (int c = 0; c < 4; ++c) {
      v16bf b = lds_load_bt(Bs + c * 16 * 40, 40, 0, lane);
      acc[c] = WMMA_BF16(a, b, acc[c]);
    }
  }

  // Epilogue. Tiles are 64-row aligned and 2048 | 64, so the batch index and
  // the s-range are uniform over the wave's strip; which/h are uniform per
  // 16-col subtile. V's transposed layout makes the 8 rows contiguous in s.
  const int n = lane & 15, half = lane >> 4;
  const int bb = mbase >> 11;                       // batch index (uniform)
  const int sb = (mbase & 2047) + wave * 16 + half * 8;  // first s of this lane
  #pragma unroll
  for (int c = 0; c < 4; ++c) {
    int gcol = nbase + c * 16 + n;
    int which = gcol >> 9;       // 0=q 1=k 2=v  (wave-uniform)
    int inner = gcol & 511;
    int h = inner >> 6, d = inner & 63;
    if (which == 2) {
      uint4 pk;
      pk.x = pack2(acc[c][0], acc[c][1]);
      pk.y = pack2(acc[c][2], acc[c][3]);
      pk.z = pack2(acc[c][4], acc[c][5]);
      pk.w = pack2(acc[c][6], acc[c][7]);
      *(uint4*)&vbuf[((size_t)(bb * CH + h) * CDH + d) * CS + sb] = pk;
    } else {
      float scale = (which == 0) ? 0.125f : 1.0f;
      __bf16* base = ((which == 0) ? qbuf : kbuf) +
                     ((size_t)(bb * CH + h) * CS + sb) * CDH + d;
      #pragma unroll
      for (int r = 0; r < 8; ++r) base[(size_t)r * CDH] = (__bf16)(acc[c][r] * scale);
    }
  }
}

// ---------------------------------------------------------------------------
// Kernel 2: flash attention per (b,h), 64 q-rows per block, streaming over keys
// S^T = K.Q^T so softmax over keys is lane-local; O^T = V^T.P^T.
// K/V tiles double-buffered via async global->LDS: prefetch chunk kc+1 while
// computing chunk kc, waiting only asynccnt<=8 (8 copies/thread/chunk).
// ---------------------------------------------------------------------------
__global__ __launch_bounds__(128) void flash_attn_kernel(
    const __bf16* __restrict__ qbuf, const __bf16* __restrict__ kbuf,
    const __bf16* __restrict__ vbuf, __bf16* __restrict__ obuf) {
  __shared__ __align__(16) __bf16 Qs[64 * 72];
  __shared__ __align__(16) __bf16 Ks[2][64 * 72];
  __shared__ __align__(16) __bf16 Vs[2][64 * 72];
  __shared__ __align__(16) __bf16 Ps[64 * 72];
  const int tid = threadIdx.x, lane = tid & 31, wave = tid >> 5;
  const int half = lane >> 4;
  const int q0 = blockIdx.x * 64;
  const int bh = blockIdx.y;
  const int qlocal = wave * 16 + (lane & 15);

  const __bf16* ksrc_base = kbuf + (size_t)bh * CS * CDH;
  const __bf16* vsrc_base = vbuf + (size_t)bh * CDH * CS;

  // Per-thread tile-copy coordinates (8 b128 transfers per K/V chunk).
  const int crow = tid >> 3;            // 0..15  (x4 rows via +16*i)
  const int ccol = (tid & 7) * 8;       // 0..56

  // Prologue: Q tile [q][dh], then chunk 0 of K/V into buffer 0.
  const __bf16* qsrc = qbuf + ((size_t)bh * CS + q0) * CDH;
  #pragma unroll
  for (int i = 0; i < 4; ++i) {
    int row = crow + i * 16;
    async_copy_b128(qsrc + row * CDH + ccol, &Qs[row * 72 + ccol]);
  }
  #pragma unroll
  for (int i = 0; i < 4; ++i) {
    int row = crow + i * 16;
    async_copy_b128(ksrc_base + (size_t)row * CDH + ccol, &Ks[0][row * 72 + ccol]);
    async_copy_b128(vsrc_base + (size_t)row * CS + ccol, &Vs[0][row * 72 + ccol]);
  }

  const v8f vzero = {0, 0, 0, 0, 0, 0, 0, 0};
  v8f oacc[4];
  #pragma unroll
  for (int dt = 0; dt < 4; ++dt) oacc[dt] = vzero;
  float m_run = -__builtin_inff(), l_run = 0.f;

  for (int kc = 0; kc < CS / 64; ++kc) {
    const int cur = kc & 1;
    if (kc + 1 < CS / 64) {
      const int key1 = (kc + 1) * 64;
      const __bf16* ksrc = ksrc_base + (size_t)key1 * CDH;
      #pragma unroll
      for (int i = 0; i < 4; ++i) {
        int row = crow + i * 16;
        async_copy_b128(ksrc + (size_t)row * CDH + ccol, &Ks[cur ^ 1][row * 72 + ccol]);
        async_copy_b128(vsrc_base + (size_t)row * CS + key1 + ccol,
                        &Vs[cur ^ 1][row * 72 + ccol]);
      }
      async_wait8();   // chunk kc (and Q) complete; chunk kc+1 stays in flight
    } else {
      async_wait0();
    }
    __syncthreads();

    // S^T tiles: 4 key-subtiles x (K-dim dh=64 -> two x32 WMMAs)
    v8f st[4];
    #pragma unroll
    for (int kt = 0; kt < 4; ++kt) {
      st[kt] = vzero;
      v16bf a0 = lds_load_a(Ks[cur] + kt * 16 * 72, 72, 0, lane);
      v16bf b0 = lds_load_bt(Qs + wave * 16 * 72, 72, 0, lane);
      st[kt] = WMMA_BF16(a0, b0, st[kt]);
      v16bf a1 = lds_load_a(Ks[cur] + kt * 16 * 72, 72, 32, lane);
      v16bf b1 = lds_load_bt(Qs + wave * 16 * 72, 72, 32, lane);
      st[kt] = WMMA_BF16(a1, b1, st[kt]);
    }

    // streaming softmax (per lane = per q column; merge halves via shfl_xor 16)
    float mx = -__builtin_inff();
    #pragma unroll
    for (int kt = 0; kt < 4; ++kt)
      #pragma unroll
      for (int r = 0; r < 8; ++r) mx = fmaxf(mx, st[kt][r]);
    mx = fmaxf(mx, __shfl_xor(mx, 16, 32));
    float mnew = fmaxf(m_run, mx);
    float corr = __expf(m_run - mnew);
    float lsum = 0.f;
    #pragma unroll
    for (int kt = 0; kt < 4; ++kt)
      #pragma unroll
      for (int r = 0; r < 8; ++r) {
        float p = __expf(st[kt][r] - mnew);
        st[kt][r] = p;
        lsum += p;
      }
    lsum += __shfl_xor(lsum, 16, 32);
    l_run = l_run * corr + lsum;
    m_run = mnew;
    #pragma unroll
    for (int dt = 0; dt < 4; ++dt) oacc[dt] *= corr;

    // P^T accumulators -> LDS as [q][key] (transposed-B layout for V^T.P^T)
    #pragma unroll
    for (int kt = 0; kt < 4; ++kt) {
      uint4 pk;
      pk.x = pack2(st[kt][0], st[kt][1]);
      pk.y = pack2(st[kt][2], st[kt][3]);
      pk.z = pack2(st[kt][4], st[kt][5]);
      pk.w = pack2(st[kt][6], st[kt][7]);
      *(uint4*)&Ps[qlocal * 72 + kt * 16 + half * 8] = pk;
    }

    // O^T += V^T . P^T : 4 dh-subtiles x (K-dim key=64 -> two x32 WMMAs)
    #pragma unroll
    for (int dt = 0; dt < 4; ++dt) {
      v16bf a0 = lds_load_a(Vs[cur] + dt * 16 * 72, 72, 0, lane);
      v16bf b0 = lds_load_bt(Ps + wave * 16 * 72, 72, 0, lane);
      oacc[dt] = WMMA_BF16(a0, b0, oacc[dt]);
      v16bf a1 = lds_load_a(Vs[cur] + dt * 16 * 72, 72, 32, lane);
      v16bf b1 = lds_load_bt(Ps + wave * 16 * 72, 72, 32, lane);
      oacc[dt] = WMMA_BF16(a1, b1, oacc[dt]);
    }
    __syncthreads();   // all waves done with buf[cur] before it is refilled
  }

  // finalize: O[q][h*64+dh] = O^T[dh][q] / l[q]   (8 contiguous bf16 per store)
  float inv = 1.0f / l_run;
  const int bidx = bh >> 3, h = bh & 7;
  __bf16* orow = obuf + ((size_t)bidx * CS + q0 + qlocal) * CINNER + h * CDH;
  #pragma unroll
  for (int dt = 0; dt < 4; ++dt) {
    uint4 pk;
    pk.x = pack2(oacc[dt][0] * inv, oacc[dt][1] * inv);
    pk.y = pack2(oacc[dt][2] * inv, oacc[dt][3] * inv);
    pk.z = pack2(oacc[dt][4] * inv, oacc[dt][5] * inv);
    pk.w = pack2(oacc[dt][6] * inv, oacc[dt][7] * inv);
    *(uint4*)(orow + dt * 16 + half * 8) = pk;
  }
}

// ---------------------------------------------------------------------------
// Kernel 3: out = O @ W_out + b_out  (fp32 output); A-tile staged async.
// ---------------------------------------------------------------------------
__global__ __launch_bounds__(128) void out_proj_kernel(
    const __bf16* __restrict__ obuf, const float* __restrict__ Wout,
    const float* __restrict__ bout, float* __restrict__ out) {
  __shared__ __align__(16) __bf16 As[64 * 40];
  __shared__ __align__(16) __bf16 Bs[64 * 40];
  const int tid = threadIdx.x, lane = tid & 31, wave = tid >> 5;
  const int nbase = blockIdx.x * 64, mbase = blockIdx.y * 64;

  const v8f vzero = {0, 0, 0, 0, 0, 0, 0, 0};
  v8f acc[4];
  #pragma unroll
  for (int c = 0; c < 4; ++c) acc[c] = vzero;

  for (int kc = 0; kc < CINNER / 32; ++kc) {
    const int k0 = kc * 32;
    __syncthreads();
    // A tile: O already bf16 -> async copy straight into LDS
    #pragma unroll
    for (int i = 0; i < 2; ++i) {
      int f4 = tid + i * 128;
      int row = f4 >> 2, c = (f4 & 3) * 8;
      async_copy_b128(obuf + (size_t)(mbase + row) * CINNER + k0 + c, &As[row * 40 + c]);
    }
    // B tile: W_out fp32 -> bf16 transposed [n][k]
    #pragma unroll
    for (int i = 0; i < 4; ++i) {
      int f4 = tid + i * 128;
      int kr = f4 >> 4, n4 = (f4 & 15) * 4;
      float4 v = *(const float4*)(Wout + (size_t)(k0 + kr) * CD + nbase + n4);
      Bs[(n4 + 0) * 40 + kr] = (__bf16)v.x;
      Bs[(n4 + 1) * 40 + kr] = (__bf16)v.y;
      Bs[(n4 + 2) * 40 + kr] = (__bf16)v.z;
      Bs[(n4 + 3) * 40 + kr] = (__bf16)v.w;
    }
    async_wait0();
    __syncthreads();
    v16bf a = lds_load_a(As + wave * 16 * 40, 40, 0, lane);
    #pragma unroll
    for (int c = 0; c < 4; ++c) {
      v16bf b = lds_load_bt(Bs + c * 16 * 40, 40, 0, lane);
      acc[c] = WMMA_BF16(a, b, acc[c]);
    }
  }

  const int n = lane & 15, half = lane >> 4;
  #pragma unroll
  for (int c = 0; c < 4; ++c) {
    int gcol = nbase + c * 16 + n;
    float bias = bout[gcol];
    #pragma unroll
    for (int r = 0; r < 8; ++r) {
      int grow = mbase + wave * 16 + half * 8 + r;
      out[(size_t)grow * CD + gcol] = acc[c][r] + bias;
    }
  }
}

// ---------------------------------------------------------------------------
extern "C" void kernel_launch(void* const* d_in, const int* in_sizes, int n_in,
                              void* d_out, int out_size, void* d_ws, size_t ws_size,
                              hipStream_t stream) {
  const float* x    = (const float*)d_in[0];
  const float* Wqkv = (const float*)d_in[1];
  const float* Wout = (const float*)d_in[2];
  const float* bout = (const float*)d_in[3];
  float* out = (float*)d_out;

  char* ws = (char*)d_ws;
  const size_t SEG = (size_t)CB * CH * CS * CDH * 2;  // 8 MiB per bf16 buffer
  __bf16* qbuf = (__bf16*)(ws + 0 * SEG);
  __bf16* kbuf = (__bf16*)(ws + 1 * SEG);
  __bf16* vbuf = (__bf16*)(ws + 2 * SEG);
  __bf16* obuf = (__bf16*)(ws + 3 * SEG);

  qkv_proj_kernel<<<dim3((3 * CINNER) / 64, (CB * CS) / 64), 128, 0, stream>>>(
      x, Wqkv, qbuf, kbuf, vbuf);
  flash_attn_kernel<<<dim3(CS / 64, CB * CH), 128, 0, stream>>>(
      qbuf, kbuf, vbuf, obuf);
  out_proj_kernel<<<dim3(CD / 64, (CB * CS) / 64), 128, 0, stream>>>(
      obuf, Wout, bout, out);
}
